// IntegratedQDropHQGCModel_69037304316296
// MI455X (gfx1250) — matrix operand order repaired and checked by reading, not compiled, not run on hardware
//
#include <hip/hip_runtime.h>

// CDNA5 / gfx1250 fused kernel:
//   out = softmax(relu(Q(relu(X@W1+b1), qw)@W2+b2)@W3+b3)
// GEMM1 done with v_wmma_f32_16x16x32_bf16 using hi/lo bf16 split of fp32
// operands (x*w ~= hi*whi + lo*whi + hi*wlo) for fp32-class accuracy.
// Memory-bound: 205.5 MB input stream -> ~8.8 us floor @ 23.3 TB/s.

typedef __attribute__((ext_vector_type(16))) __bf16 v16bf;
typedef __attribute__((ext_vector_type(8)))  float  v8f;

#define NQ     4
#define NL     2
#define KDIM   784
#define NCHUNK 25      // ceil(784/32); chunk 24 has only K=768..783 valid

// ---- 4-qubit statevector helpers (fully unrolled; state lives in VGPRs) ----
// index i = q0*8 + q1*4 + q2*2 + q3 ; wire q <-> mask (8 >> q)
template<int MASK>
__device__ __forceinline__ void rx_apply(float re[16], float im[16],
                                         float ch, float sh) {
#pragma unroll
  for (int i = 0; i < 16; ++i)
    if (!(i & MASK)) {
      const int j = i | MASK;
      float r0 = re[i], i0 = im[i], r1 = re[j], i1 = im[j];
      re[i] = ch * r0 + sh * i1;
      im[i] = ch * i0 - sh * r1;
      re[j] = ch * r1 + sh * i0;
      im[j] = ch * i1 - sh * r0;
    }
}

template<int CM, int TM>
__device__ __forceinline__ void cnot_apply(float re[16], float im[16]) {
#pragma unroll
  for (int i = 0; i < 16; ++i)
    if ((i & CM) && !(i & TM)) {
      const int j = i | TM;
      float t;
      t = re[i]; re[i] = re[j]; re[j] = t;
      t = im[i]; im[i] = im[j]; im[j] = t;
    }
}

__global__ __launch_bounds__(256) void fused_qmlp(
    const float* __restrict__ X,    // [B,784]
    const float* __restrict__ W1,   // [784,4]
    const float* __restrict__ b1,   // [4]
    const float* __restrict__ qw,   // [2,4]
    const float* __restrict__ W2,   // [4,32]
    const float* __restrict__ b2,   // [32]
    const float* __restrict__ W3,   // [32,2]
    const float* __restrict__ b3,   // [2]
    float* __restrict__ Out,        // [B,2]
    int Bsz)
{
  // W1 (split hi/lo) staged in WMMA B-fragment order:
  // entry (c, kh, n, j) <-> K = c*32 + kh*8 + (j&7) + (j>>3)*16, column n (n>=4 zero-padded)
  __shared__ alignas(32) __bf16 Bhi[NCHUNK * 2 * 16 * 16];
  __shared__ alignas(32) __bf16 Blo[NCHUNK * 2 * 16 * 16];
  __shared__ float preLds[256 * NQ];   // relu(X@W1+b1) for this block

  const int tid  = threadIdx.x;
  const int lane = tid & 31;
  const int w    = tid >> 5;           // wave id 0..7, owns 32 samples

  // ---- stage split-bf16 padded W1 into LDS (once per block) ----
  for (int i = tid; i < NCHUNK * 2 * 16 * 16; i += 256) {
    int j  = i & 15;
    int n  = (i >> 4) & 15;
    int ck = i >> 8;                              // c*2 + kh
    int k  = (ck >> 1) * 32 + (ck & 1) * 8 + (j & 7) + (j >> 3) * 16;  // K index
    float wv = (n < NQ && k < KDIM) ? W1[k * NQ + n] : 0.f;
    __bf16 hi = (__bf16)wv;
    Bhi[i] = hi;
    Blo[i] = (__bf16)(wv - (float)hi);
  }
  __syncthreads();

  const int r  = lane & 15;   // row-in-tile for A, column n for B/C
  const int kh = lane >> 4;   // K-half selector per ISA fragment layout

  // ---- GEMM1: two 16-sample tiles per wave ----
  for (int t = 0; t < 2; ++t) {
    int rowg = blockIdx.x * 256 + w * 32 + t * 16 + r;
    if (rowg >= Bsz) rowg = Bsz - 1;              // keep EXEC full for WMMA
    const float* rowp = X + (size_t)rowg * KDIM;
    v8f acc = {};

    for (int c = 0; c < NCHUNK; ++c) {
      // A fragment: v16bf elem e<8 -> K = kh*8+e ; e>=8 -> K = 16+kh*8+(e-8)
      const float* p = rowp + c * 32 + kh * 8;
      float4 a0 = *(const float4*)(p);
      float4 a1 = *(const float4*)(p + 4);
      float4 a2, a3;
      if (c < NCHUNK - 1) {
        a2 = *(const float4*)(p + 16);
        a3 = *(const float4*)(p + 20);
      } else {                                    // K>=784 zero-padded
        a2 = make_float4(0.f, 0.f, 0.f, 0.f);
        a3 = a2;
      }
      float av[16] = {a0.x, a0.y, a0.z, a0.w, a1.x, a1.y, a1.z, a1.w,
                      a2.x, a2.y, a2.z, a2.w, a3.x, a3.y, a3.z, a3.w};
      v16bf ahi, alo;
#pragma unroll
      for (int e = 0; e < 16; ++e) {
        __bf16 h = (__bf16)av[e];
        ahi[e] = h;
        alo[e] = (__bf16)(av[e] - (float)h);
      }

      // B fragment: lane holds column n=r, K = kh*8 + 0..7, 16 + kh*8 + 0..7 (contiguous 32B)
      int off = ((c * 2 + kh) * 16 + r) * 16;
      v16bf bh = *(const v16bf*)&Bhi[off];
      v16bf bl = *(const v16bf*)&Blo[off];

      // x*w ~= hi*whi + lo*whi + hi*wlo   (drop lo*wlo, ~2^-16 rel err)
      acc = __builtin_amdgcn_wmma_f32_16x16x32_bf16(false, ahi, false, bh,
                                                    (short)0, acc, false, false);
      acc = __builtin_amdgcn_wmma_f32_16x16x32_bf16(false, alo, false, bh,
                                                    (short)0, acc, false, false);
      acc = __builtin_amdgcn_wmma_f32_16x16x32_bf16(false, ahi, false, bl,
                                                    (short)0, acc, false, false);
    }

    // C layout: vgpr i, lane -> sample m = kh*8+i, column n = r
    if (r < NQ) {
      float bias = b1[r];
#pragma unroll
      for (int i = 0; i < 8; ++i) {
        int m = kh * 8 + i;
        float v = acc[i] + bias;
        v = v > 0.f ? v : 0.f;                    // relu
        preLds[(w * 32 + t * 16 + m) * NQ + r] = v;
      }
    }
  }
  __syncthreads();

  // ---- quantum circuit: one sample per lane (all 32 lanes busy) ----
  float ang[NQ];
#pragma unroll
  for (int q = 0; q < NQ; ++q) ang[q] = preLds[tid * NQ + q];

  float cg[(1 + NL) * NQ], sg[(1 + NL) * NQ];
#pragma unroll
  for (int q = 0; q < NQ; ++q) {
    float h = 0.5f * ang[q];
    cg[q] = __cosf(h); sg[q] = __sinf(h);
  }
#pragma unroll
  for (int l = 0; l < NL; ++l)
#pragma unroll
    for (int q = 0; q < NQ; ++q) {
      float h = 0.5f * qw[l * NQ + q];            // uniform -> scalar loads
      cg[(1 + l) * NQ + q] = __cosf(h);
      sg[(1 + l) * NQ + q] = __sinf(h);
    }

  float re[16], im[16];
#pragma unroll
  for (int i = 0; i < 16; ++i) { re[i] = 0.f; im[i] = 0.f; }
  re[0] = 1.f;

  // AngleEmbedding: RX(x_q) on wire q
  rx_apply<8>(re, im, cg[0], sg[0]);
  rx_apply<4>(re, im, cg[1], sg[1]);
  rx_apply<2>(re, im, cg[2], sg[2]);
  rx_apply<1>(re, im, cg[3], sg[3]);

  // BasicEntanglerLayers: RX(w[l,q]) then CNOT ring 0->1->2->3->0
#pragma unroll
  for (int l = 0; l < NL; ++l) {
    const int b = (1 + l) * NQ;
    rx_apply<8>(re, im, cg[b + 0], sg[b + 0]);
    rx_apply<4>(re, im, cg[b + 1], sg[b + 1]);
    rx_apply<2>(re, im, cg[b + 2], sg[b + 2]);
    rx_apply<1>(re, im, cg[b + 3], sg[b + 3]);
    cnot_apply<8, 4>(re, im);
    cnot_apply<4, 2>(re, im);
    cnot_apply<2, 1>(re, im);
    cnot_apply<1, 8>(re, im);
  }

  // <Z_q> = sum_i p(i) * (+1 if bit q == 0 else -1), then NaN scrub
  float z[NQ];
#pragma unroll
  for (int q = 0; q < NQ; ++q) {
    const int m = 8 >> q;
    float a = 0.f;
#pragma unroll
    for (int i = 0; i < 16; ++i) {
      float p = re[i] * re[i] + im[i] * im[i];
      a += (i & m) ? -p : p;
    }
    z[q] = (a == a) ? a : 0.f;
  }

  // ---- MLP head: relu(z@W2+b2) @ W3 + b3, softmax over 2 classes ----
  float logit0 = b3[0], logit1 = b3[1];
#pragma unroll
  for (int j = 0; j < 32; ++j) {
    float a2 = b2[j];
#pragma unroll
    for (int q = 0; q < NQ; ++q) a2 = fmaf(z[q], W2[q * 32 + j], a2);
    a2 = a2 > 0.f ? a2 : 0.f;
    logit0 = fmaf(a2, W3[j * 2 + 0], logit0);
    logit1 = fmaf(a2, W3[j * 2 + 1], logit1);
  }
  float mx = fmaxf(logit0, logit1);
  float e0 = __expf(logit0 - mx);
  float e1 = __expf(logit1 - mx);
  float inv = 1.f / (e0 + e1);

  int S = blockIdx.x * 256 + tid;
  if (S < Bsz) {
    float2 o; o.x = e0 * inv; o.y = e1 * inv;
    *(float2*)&Out[S * 2] = o;
  }
}

extern "C" void kernel_launch(void* const* d_in, const int* in_sizes, int n_in,
                              void* d_out, int out_size, void* d_ws, size_t ws_size,
                              hipStream_t stream) {
  const float* X  = (const float*)d_in[0];
  const float* W1 = (const float*)d_in[1];
  const float* b1 = (const float*)d_in[2];
  const float* qw = (const float*)d_in[3];
  const float* W2 = (const float*)d_in[4];
  const float* b2 = (const float*)d_in[5];
  const float* W3 = (const float*)d_in[6];
  const float* b3 = (const float*)d_in[7];
  float* Out = (float*)d_out;

  int Bsz    = in_sizes[0] / KDIM;       // 65536
  int blocks = (Bsz + 255) / 256;        // 256 samples per block

  hipLaunchKernelGGL(fused_qmlp, dim3(blocks), dim3(256), 0, stream,
                     X, W1, b1, qw, W2, b2, W3, b3, Out, Bsz);
}
